// Retain_pertime_11235634446890
// MI455X (gfx1250) — matrix operand order repaired and checked by reading, not compiled, not run on hardware
//
#include <hip/hip_runtime.h>
#include <hip/hip_bf16.h>

typedef __attribute__((ext_vector_type(16))) _Float16 v16h;
typedef __attribute__((ext_vector_type(8)))  _Float16 v8h;
typedef __attribute__((ext_vector_type(8)))  float    v8f;

// ---------------------------------------------------------------------------
// Fragment loaders for V_WMMA_F32_16X16X32_F16 (wave32).
// A (16x32, row-major source, ld = K):
//   lane L (row = L%16): halves 0..7 = K [c..c+7], halves 8..15 = K [c+16..c+23],
//   where c = 8*(L>=16).  (ISA 7.12.2 "16-bit A-Matrix 16x32")
// B (32x16) built from a row-major [N x K] weight W (B[k][n] = W[n][k]):
//   lane L (col = L%16): same K order as A: halves 0..7 = K [c..c+7], halves 8..15 = K [c+16..c+23].
// ---------------------------------------------------------------------------
__device__ __forceinline__ v16h ld_fragA(const _Float16* base, int ld, int k0, int lane) {
    const int r    = lane & 15;
    const int coff = (lane >> 4) << 3;            // 0 or 8
    const _Float16* p = base + r * ld + k0 + coff;
    v8h lo = *(const v8h*)(p);
    v8h hi = *(const v8h*)(p + 16);
    v16h f;
#pragma unroll
    for (int i = 0; i < 8; ++i) { f[i] = lo[i]; f[i + 8] = hi[i]; }
    return f;
}

__device__ __forceinline__ v16h ld_fragB(const _Float16* base, int ld, int k0, int lane) {
    const int n    = lane & 15;
    const int koff = (lane >> 4) << 3;            // 0 or 8
    const _Float16* p = base + n * ld + k0 + koff;
    v8h lo = *(const v8h*)(p);
    v8h hi = *(const v8h*)(p + 16);
    v16h f;
#pragma unroll
    for (int i = 0; i < 8; ++i) { f[i] = lo[i]; f[i + 8] = hi[i]; }
    return f;
}

// One 16x16 output tile: C[m0:m0+16, n0:n0+16] = A[m0:,:K] * W[n0:,:K]^T (+post)
// MODE 0: raw   MODE 1: + bias[n]   MODE 2: tanh(scale*acc + bias[n])
template <int MODE>
__device__ __forceinline__ void wmma_tile(const _Float16* A, int lda,
                                          const _Float16* W, int ldw,
                                          float* C, int ldc, const float* bias,
                                          int m0, int n0, int K, int lane, float scale) {
    v8f acc = {};
    for (int k0 = 0; k0 < K; k0 += 32) {
        v16h a = ld_fragA(A + m0 * lda, lda, k0, lane);
        v16h b = ld_fragB(W + n0 * ldw, ldw, k0, lane);
        acc = __builtin_amdgcn_wmma_f32_16x16x32_f16(false, a, false, b,
                                                     (short)0, acc, false, false);
    }
    const int n  = n0 + (lane & 15);
    const int mb = m0 + ((lane >> 4) << 3);
#pragma unroll
    for (int r = 0; r < 8; ++r) {
        float v = acc[r];
        if (MODE == 1) v += bias[n];
        if (MODE == 2) v = tanhf(scale * v + bias[n]);
        C[(mb + r) * ldc + n] = v;
    }
}

// ---------------------------------------------------------------------------
// Model constants (fixed by the reference file)
// ---------------------------------------------------------------------------
#define TB   64
#define TT   64
#define TD   64
#define TE   128
#define TH   128
#define TG   384            // 3*H
#define TP   63             // T-1 prefixes

// ---------------------------------------------------------------------------
// prep: f32 -> f16 weights, + W_emb transpose
// ---------------------------------------------------------------------------
__global__ void prep_weights(const float* WiA, const float* WhA,
                             const float* WiB, const float* WhB,
                             const float* Wb,  const float* Wemb,
                             _Float16* WiA16, _Float16* WhA16,
                             _Float16* WiB16, _Float16* WhB16,
                             _Float16* Wb16,  _Float16* WembT16) {
    const int i  = blockIdx.x * blockDim.x + threadIdx.x;
    const int st = gridDim.x * blockDim.x;
    for (int k = i; k < TG * TE; k += st) {
        WiA16[k] = (_Float16)WiA[k];
        WhA16[k] = (_Float16)WhA[k];
        WiB16[k] = (_Float16)WiB[k];
        WhB16[k] = (_Float16)WhB[k];
    }
    for (int k = i; k < TE * TH; k += st) Wb16[k] = (_Float16)Wb[k];
    for (int k = i; k < TD * TE; k += st) {
        int d = k >> 7, e = k & 127;                 // WembT[d][e] = Wemb[e][d]
        WembT16[k] = (_Float16)Wemb[e * TD + d];
    }
}

// emb[t,b,e] = x[b,t,:] . Wemb[e,:] + bemb[e]
__global__ void emb_kernel(const float* x, const float* Wemb, const float* bemb,
                           float* emb32, _Float16* emb16) {
    const int idx = blockIdx.x * blockDim.x + threadIdx.x;   // T*B*E = 524288
    if (idx >= TT * TB * TE) return;
    const int e = idx & 127, row = idx >> 7;
    const int t = row >> 6, b = row & 63;
    const float* xp = x + (b * TT + t) * TD;
    const float* wp = Wemb + e * TD;
    float acc = bemb[e];
#pragma unroll 8
    for (int d = 0; d < TD; ++d) acc += xp[d] * wp[d];
    emb32[idx] = acc;
    emb16[idx] = (_Float16)acc;
}

// gi = emb16 @ W^T + bias : one 16x16 tile per wave
__global__ void gi_kernel(const _Float16* A, const _Float16* W, const float* bias,
                          float* out, int Mtiles, int Ntiles, int K, int ldc) {
    const int lane = threadIdx.x & 31;
    const int wave = blockIdx.x * (blockDim.x >> 5) + (threadIdx.x >> 5);
    if (wave >= Mtiles * Ntiles) return;
    const int m0 = (wave / Ntiles) * 16;
    const int n0 = (wave % Ntiles) * 16;
    wmma_tile<1>(A, K, W, K, out, ldc, bias, m0, n0, K, lane, 0.f);
}

__device__ __forceinline__ void gru_update(float* h, _Float16* h16, const float* gh,
                                           const float* gi, const float* bh, int tid) {
    for (int i = tid; i < TB * TH; i += 256) {
        const int b = i >> 7, c = i & 127;
        const float* gir = gi + b * TG;
        const float* ghr = gh + b * TG;
        float r = 1.f / (1.f + __expf(-(gir[c]        + ghr[c]        + bh[c])));
        float z = 1.f / (1.f + __expf(-(gir[c + 128]  + ghr[c + 128]  + bh[c + 128])));
        float n = tanhf(gir[c + 256] + r * (ghr[c + 256] + bh[c + 256]));
        float hn = (1.f - z) * n + z * h[i];
        h[i]  = hn;
        h16[i] = (_Float16)hn;
    }
}

// ---------------------------------------------------------------------------
// Main: one block per prefix p. Hidden state + accumulators live in LDS.
// Online (flash-style) softmax over scan steps s.
// ---------------------------------------------------------------------------
__global__ __launch_bounds__(256) void retain_kernel(
    const float* __restrict__ x, const float* __restrict__ emb32,
    const float* __restrict__ giA, const float* __restrict__ giB,
    const _Float16* __restrict__ WhA16, const _Float16* __restrict__ WhB16,
    const _Float16* __restrict__ Wb16,  const _Float16* __restrict__ WembT16,
    const float* __restrict__ bhA, const float* __restrict__ bhB,
    const float* __restrict__ Wa, const float* __restrict__ ba,
    const float* __restrict__ bb, const float* __restrict__ Wo,
    const float* __restrict__ bo, const int* __restrict__ shortp,
    float* __restrict__ out) {

    const int p  = blockIdx.x;
    const int sh = *shortp;
    if (p <= sh) return;                 // this prefix produces no output

    extern __shared__ char smem[];
    float* gh   = (float*)smem;                      // 64*384
    float* hA   = gh   + TB * TG;                    // 64*128
    float* hB   = hA   + TB * TH;
    float* cacc = hB   + TB * TH;                    // 64*128
    float* gacc = cacc + TB * TE;                    // 64*64
    float* preA = gacc + TB * TD;                    // 64
    float* mbuf = preA + 64;
    float* lbuf = mbuf + 64;
    float* scb  = lbuf + 64;
    float* wgt  = scb  + 64;
    _Float16* hA16 = (_Float16*)(wgt + 64);          // 64*128
    _Float16* hB16 = hA16 + TB * TH;
    _Float16* Mt16 = hB16 + TB * TH;
    float* betaS = gh;                               // alias: 64*128 (gh free then)
    float* echS  = gh + TB * TE;                     // alias: 64*64

    const int tid  = threadIdx.x;
    const int lane = tid & 31;
    const int wave = tid >> 5;

    for (int i = tid; i < TB * TH; i += 256) {
        hA[i] = 0.f; hB[i] = 0.f; cacc[i] = 0.f;
        hA16[i] = (_Float16)0.f; hB16[i] = (_Float16)0.f;
    }
    for (int i = tid; i < TB * TD; i += 256) gacc[i] = 0.f;
    if (tid < 64) { mbuf[tid] = -__builtin_inff(); lbuf[tid] = 0.f; }
    __syncthreads();

    for (int s = 0; s <= p; ++s) {
        const int j = p - s;

        // ---- GRU A: gh = hA @ WhA^T  (4x24 tiles over 8 waves) ----
        for (int t = wave; t < 4 * 24; t += 8)
            wmma_tile<0>(hA16, TH, WhA16, TH, gh, TG, nullptr,
                         (t / 24) * 16, (t % 24) * 16, TH, lane, 0.f);
        __syncthreads();
        gru_update(hA, hA16, gh, giA + (size_t)(j * TB) * TG, bhA, tid);
        __syncthreads();

        // ---- GRU B ----
        for (int t = wave; t < 4 * 24; t += 8)
            wmma_tile<0>(hB16, TH, WhB16, TH, gh, TG, nullptr,
                         (t / 24) * 16, (t % 24) * 16, TH, lane, 0.f);
        __syncthreads();
        gru_update(hB, hB16, gh, giB + (size_t)(j * TB) * TG, bhB, tid);
        __syncthreads();

        // ---- preA[b] = 0.5*hA[b,:].Wa + ba ----
        if (tid < 64) {
            const float* hp = hA + tid * TH;
            float acc = ba[0];
#pragma unroll 8
            for (int k = 0; k < TH; ++k) acc += 0.5f * hp[k] * Wa[k];
            preA[tid] = acc;
        }
        // ---- beta = tanh(0.5 * hB @ Wb^T + bb)  (4x8 tiles) ----
        for (int t = wave; t < 4 * 8; t += 8)
            wmma_tile<2>(hB16, TH, Wb16, TH, betaS, TE, bb,
                         (t / 8) * 16, (t % 8) * 16, TH, lane, 0.5f);
        __syncthreads();

        // ---- Mt = beta * Wo (f16 A operand) ----
        for (int i = tid; i < TB * TE; i += 256)
            Mt16[i] = (_Float16)(betaS[i] * Wo[i & 127]);
        __syncthreads();

        // ---- e_chain = Mt @ W_emb  (4x4 tiles, B from W_emb^T) ----
        for (int t = wave; t < 4 * 4; t += 8)
            wmma_tile<0>(Mt16, TE, WembT16, TE, echS, TD, nullptr,
                         (t / 4) * 16, (t % 4) * 16, TE, lane, 0.f);

        // ---- online softmax scalars ----
        if (tid < 64) {
            float pre = preA[tid], mo = mbuf[tid];
            float mn = fmaxf(mo, pre);
            float sc = __expf(mo - mn), w = __expf(pre - mn);
            scb[tid] = sc; wgt[tid] = w;
            lbuf[tid] = lbuf[tid] * sc + w;
            mbuf[tid] = mn;
        }
        __syncthreads();

        // ---- accumulate context & interpretability weights ----
        const float* eg = emb32 + (size_t)(j * TB) * TE;
        for (int i = tid; i < TB * TE; i += 256) {
            const int b = i >> 7;
            cacc[i] = cacc[i] * scb[b] + wgt[b] * betaS[i] * eg[i];
        }
        for (int i = tid; i < TB * TD; i += 256) {
            const int b = i >> 6, d = i & 63;
            gacc[i] = gacc[i] * scb[b] + wgt[b] * echS[i] * x[(b * TT + j) * TD + d];
        }
        __syncthreads();
    }

    // ---- finalize: pred and weight outputs ----
    const int Pout = (TT - 2) - sh;          // 60 for sh==2
    const int pcol = p - sh - 1;
    if (tid < 64) {
        const float* cp = cacc + tid * TE;
        const float inv = 1.f / lbuf[tid];
        float acc = bo[0];
#pragma unroll 8
        for (int e = 0; e < TE; ++e) acc += cp[e] * inv * Wo[e];
        out[tid * Pout + pcol] = acc;        // pred [B, Pout, 1]
    }
    const float invP = 1.f / (float)(p + 1);
    for (int i = tid; i < TB * TD; i += 256) {
        const int b = i >> 6, d = i & 63;
        out[TB * Pout + (b * Pout + pcol) * TD + d] =
            gacc[i] * invP / lbuf[b];        // weight [B, Pout, D]
    }
}

// ---------------------------------------------------------------------------
extern "C" void kernel_launch(void* const* d_in, const int* in_sizes, int n_in,
                              void* d_out, int out_size, void* d_ws, size_t ws_size,
                              hipStream_t stream) {
    (void)in_sizes; (void)n_in; (void)out_size; (void)ws_size;
    const float* x    = (const float*)d_in[0];
    const float* Wemb = (const float*)d_in[1];
    const float* bemb = (const float*)d_in[2];
    const float* WiA  = (const float*)d_in[3];
    const float* WhA  = (const float*)d_in[4];
    const float* biA  = (const float*)d_in[5];
    const float* bhA  = (const float*)d_in[6];
    const float* WiB  = (const float*)d_in[7];
    const float* WhB  = (const float*)d_in[8];
    const float* biB  = (const float*)d_in[9];
    const float* bhB  = (const float*)d_in[10];
    const float* Wa   = (const float*)d_in[11];
    const float* ba   = (const float*)d_in[12];
    const float* Wb   = (const float*)d_in[13];
    const float* bb   = (const float*)d_in[14];
    const float* Wo   = (const float*)d_in[15];
    const float* bo   = (const float*)d_in[16];
    const int*   shp  = (const int*)d_in[17];
    float* out = (float*)d_out;

    char* ws = (char*)d_ws;
    size_t off = 0;
    auto carve = [&](size_t bytes) -> void* {
        void* p = ws + off;
        off += (bytes + 255) & ~(size_t)255;
        return p;
    };
    float*    emb32   = (float*)   carve((size_t)TT * TB * TE * 4);
    _Float16* emb16   = (_Float16*)carve((size_t)TT * TB * TE * 2);
    float*    giA     = (float*)   carve((size_t)TT * TB * TG * 4);
    float*    giB     = (float*)   carve((size_t)TT * TB * TG * 4);
    _Float16* WiA16   = (_Float16*)carve((size_t)TG * TE * 2);
    _Float16* WhA16   = (_Float16*)carve((size_t)TG * TH * 2);
    _Float16* WiB16   = (_Float16*)carve((size_t)TG * TE * 2);
    _Float16* WhB16   = (_Float16*)carve((size_t)TG * TH * 2);
    _Float16* Wb16    = (_Float16*)carve((size_t)TE * TH * 2);
    _Float16* WembT16 = (_Float16*)carve((size_t)TD * TE * 2);

    prep_weights<<<dim3(64), dim3(256), 0, stream>>>(
        WiA, WhA, WiB, WhB, Wb, Wemb, WiA16, WhA16, WiB16, WhB16, Wb16, WembT16);

    emb_kernel<<<dim3((TT * TB * TE) / 256), dim3(256), 0, stream>>>(
        x, Wemb, bemb, emb32, emb16);

    // gi = emb @ Wi^T + bi : (4096/16)*(384/16) = 6144 tiles, 8 waves/block
    gi_kernel<<<dim3(768), dim3(256), 0, stream>>>(emb16, WiA16, biA, giA,
                                                   (TT * TB) / 16, TG / 16, TE, TG);
    gi_kernel<<<dim3(768), dim3(256), 0, stream>>>(emb16, WiB16, biB, giB,
                                                   (TT * TB) / 16, TG / 16, TE, TG);

    // dynamic LDS: gh + hA + hB + cacc + gacc + 5*64 f32, then 3 f16 buffers
    const size_t smem_f32 = (size_t)(TB * TG + 3 * TB * TH + TB * TD + 5 * 64) * 4;
    const size_t smem_f16 = (size_t)(3 * TB * TH) * 2;
    const size_t smem = smem_f32 + smem_f16;   // 263,424 B (< 320 KB/WGP)

    retain_kernel<<<dim3(TP), dim3(256), smem, stream>>>(
        x, emb32, giA, giB, WhA16, WhB16, Wb16, WembT16,
        bhA, bhB, Wa, ba, bb, Wo, bo, shp, out);
}
